// Attention_82583631167775
// MI455X (gfx1250) — matrix operand order repaired and checked by reading, not compiled, not run on hardware
//
#include <hip/hip_runtime.h>
#include <hip/hip_bf16.h>
#include <math.h>

typedef __bf16 bf16;
typedef __attribute__((ext_vector_type(16))) __bf16 bf16x16;
typedef __attribute__((ext_vector_type(8)))  __bf16 bf16x8;
typedef __attribute__((ext_vector_type(8)))  float   v8f;
typedef __attribute__((ext_vector_type(4)))  unsigned tdm_u32x4;
typedef __attribute__((ext_vector_type(8)))  int      tdm_i32x8;
typedef __attribute__((ext_vector_type(4)))  int      tdm_i32x4;

#define D_MODEL 2048
#define SEQ     2048
#define BATCH   2
#define N_HEADS 16
#define D_HEAD  128
#define TOKENS  (BATCH * SEQ)            /* 4096 */
#define SCALE_F 0.08838834764831845f     /* 1/sqrt(128) */

// ---------------- helpers ----------------

__device__ inline bf16 f2bf(float f) {
  unsigned u = __float_as_uint(f);
  u += 0x7FFFu + ((u >> 16) & 1u);   // round-to-nearest-even
  unsigned short h = (unsigned short)(u >> 16);
  bf16 r; __builtin_memcpy(&r, &h, sizeof(r));
  return r;
}

// A-matrix 16x32 bf16 fragment: row M = lane%16, K = half*16 + {0..15} contiguous
// (same k order as load_frag_b).
__device__ inline bf16x16 load_frag_a(const bf16* p, int lane) {
  const bf16* q = p + ((lane & 16) ? 16 : 0);
  bf16x8 lo = *(const bf16x8*)(q);
  bf16x8 hi = *(const bf16x8*)(q + 8);
  bf16x16 r;
#pragma unroll
  for (int i = 0; i < 8; ++i) { r[i] = lo[i]; r[i + 8] = hi[i]; }
  return r;
}

// B-matrix 32x16 bf16 fragment: column N = lane%16, K = half*16 + {0..15} contiguous.
// p points to a row of B^T (column of B), contiguous along K. Works for global or LDS.
__device__ inline bf16x16 load_frag_b(const bf16* p, int lane) {
  const bf16* q = p + ((lane & 16) ? 16 : 0);
  bf16x8 lo = *(const bf16x8*)(q);
  bf16x8 hi = *(const bf16x8*)(q + 8);
  bf16x16 r;
#pragma unroll
  for (int i = 0; i < 8; ++i) { r[i] = lo[i]; r[i + 8] = hi[i]; }
  return r;
}

__device__ inline v8f wmma_bf16(bf16x16 a, bf16x16 b, v8f c) {
  return __builtin_amdgcn_wmma_f32_16x16x32_bf16(
      /*neg_a=*/false, a, /*neg_b=*/false, b,
      /*c_mod=*/(short)0, c, /*reuse_a=*/false, /*reuse_b=*/false);
}

// Row-wise reductions across the 16 lanes of each wave-half (C layout puts N on lanes).
__device__ inline float hred_max(float v) {
  v = fmaxf(v, __shfl_xor(v, 8, 32));
  v = fmaxf(v, __shfl_xor(v, 4, 32));
  v = fmaxf(v, __shfl_xor(v, 2, 32));
  v = fmaxf(v, __shfl_xor(v, 1, 32));
  return v;
}
__device__ inline float hred_sum(float v) {
  v += __shfl_xor(v, 8, 32);
  v += __shfl_xor(v, 4, 32);
  v += __shfl_xor(v, 2, 32);
  v += __shfl_xor(v, 1, 32);
  return v;
}

__device__ inline void store_val(float* p, float v) { *p = v; }
__device__ inline void store_val(bf16*  p, float v) { *p = f2bf(v); }

// CDNA5 async global->LDS copy (16 bytes per lane), tracked by ASYNCcnt.
__device__ inline void async_copy_b128(unsigned lds_byte_addr, const void* gaddr) {
  unsigned long long ga = (unsigned long long)gaddr;
  asm volatile("global_load_async_to_lds_b128 %0, %1, off"
               :: "v"(lds_byte_addr), "v"(ga)
               : "memory");
}
__device__ inline void wait_async0() {
  asm volatile("s_wait_asynccnt 0x0" ::: "memory");
}

// CDNA5 Tensor Data Mover: 2D tile load (tile_rows x tile_k bf16 elems, row stride
// stride_elems) from global to LDS, packed densely at lds_addr. D# per ISA 8.3/8.4.
// clang-23 toolchain form: 6 args (g0, g1, g2, g3, g4, cpol).
__device__ inline void tdm_load_2d(unsigned lds_addr, const void* gaddr,
                                   unsigned tile_k, unsigned tile_rows,
                                   unsigned long long stride_elems) {
  unsigned long long ga = (unsigned long long)gaddr;
  tdm_u32x4 g0;
  g0[0] = 1u;                                              // count=1 valid descriptor
  g0[1] = lds_addr;                                        // LDS byte address
  g0[2] = (unsigned)(ga & 0xFFFFFFFFu);                    // global_addr[31:0]
  g0[3] = (unsigned)((ga >> 32) & 0x01FFFFFFu) | (2u << 30); // addr[56:32] | type=2

  const unsigned T0 = 1u << 20, T1 = 1u << 20;             // oversized tensor dims: no OOB clip
  tdm_i32x8 g1;
  g1[0] = (int)(1u << 16);                                 // data_size=1 (2B), wg_mask=0
  g1[1] = (int)((T0 & 0xFFFFu) << 16);                     // [63:48] = tensor_dim0[15:0]
  g1[2] = (int)(((T0 >> 16) & 0xFFFFu) | ((T1 & 0xFFFFu) << 16)); // dim0 hi16 | dim1 lo16
  g1[3] = (int)(((T1 >> 16) & 0xFFFFu) | ((tile_k & 0xFFFFu) << 16)); // dim1 hi16 | tile_dim0
  g1[4] = (int)(tile_rows & 0xFFFFu);                      // tile_dim1 (tile_dim2 = 0)
  g1[5] = (int)(unsigned)(stride_elems & 0xFFFFFFFFu);     // tensor_dim0_stride[31:0]
  g1[6] = (int)(unsigned)((stride_elems >> 32) & 0xFFFFu); // stride[47:32] (dim1_stride=0)
  g1[7] = 0;

  tdm_i32x4 g2 = {0, 0, 0, 0};                             // 2D: groups 2/3 unused
  tdm_i32x4 g3 = {0, 0, 0, 0};
  tdm_i32x8 g4 = {0, 0, 0, 0, 0, 0, 0, 0};                 // unused trailing group
  __builtin_amdgcn_tensor_load_to_lds(g0, g1, g2, g3, g4, 0);  // TENSORcnt++
}

// ---------------- conversion / transpose kernels ----------------

__global__ void k_f32_to_bf16(const float* __restrict__ in, bf16* __restrict__ out,
                              long long n) {
  long long i = (long long)blockIdx.x * blockDim.x + threadIdx.x;
  long long stride = (long long)gridDim.x * blockDim.x;
  for (; i < n; i += stride) out[i] = f2bf(in[i]);
}

// in: f32 [R][C]  ->  out: bf16 [C][R]   (R,C multiples of 32)
__global__ void k_transpose_f32_bf16(const float* __restrict__ in, bf16* __restrict__ out,
                                     int R, int C) {
  __shared__ float tile[32][33];
  int tx = threadIdx.x & 31;
  int ty = threadIdx.x >> 5;               // 0..7
  int r0 = blockIdx.y * 32;
  int c0 = blockIdx.x * 32;
#pragma unroll
  for (int i = ty; i < 32; i += 8)
    tile[i][tx] = in[(size_t)(r0 + i) * C + (c0 + tx)];
  __syncthreads();
#pragma unroll
  for (int i = ty; i < 32; i += 8)
    out[(size_t)(c0 + i) * R + (r0 + tx)] = f2bf(tile[tx][i]);
}

// batched bf16 transpose: in [b][R][C] -> out [b][C][R]
__global__ void k_transpose_bf16(const bf16* __restrict__ in, bf16* __restrict__ out,
                                 int R, int C) {
  __shared__ bf16 tile[32][33];
  int bz = blockIdx.z;
  const bf16* ib = in  + (size_t)bz * R * C;
  bf16*       ob = out + (size_t)bz * R * C;
  int tx = threadIdx.x & 31;
  int ty = threadIdx.x >> 5;
  int r0 = blockIdx.y * 32;
  int c0 = blockIdx.x * 32;
#pragma unroll
  for (int i = ty; i < 32; i += 8)
    tile[i][tx] = ib[(size_t)(r0 + i) * C + (c0 + tx)];
  __syncthreads();
#pragma unroll
  for (int i = ty; i < 32; i += 8)
    ob[(size_t)(c0 + i) * R + (r0 + tx)] = tile[tx][i];
}

// ---------------- WMMA GEMM: C[M,N] = A[M,K] * B[K,N], B given transposed ----------------
// grid: (N/64, M/128, batches), block 256 (8 waves; each wave owns a 16x64 strip).
// Per K=64 slab: the block's 128x64 A panel arrives via one TDM descriptor (wave 0),
// the shared 64x64 B strip via per-lane CDNA5 async-to-LDS copies. 8 WMMAs per slab.

template <typename OUT_T>
__global__ __launch_bounds__(256) void k_gemm_bf16(
    const bf16* __restrict__ A, const bf16* __restrict__ Bt, OUT_T* __restrict__ C,
    int K, int N, long long sA, long long sB, long long sC) {
  __shared__ __align__(16) bf16 Atile[128 * 64];   // 16KB, row stride 64 elems
  __shared__ __align__(16) bf16 Btile[64 * 64];    //  8KB, row stride 64 elems

  const int tid  = threadIdx.x;
  const int wave = tid >> 5;
  const int lane = tid & 31;
  const int hlf  = lane >> 4;
  const int lidx = lane & 15;
  const int mt = blockIdx.y * 8 + wave;
  const int nt = blockIdx.x;
  const int bz = blockIdx.z;

  const bf16* Ablk = A  + (long long)bz * sA + (size_t)(blockIdx.y * 128) * K;
  const bf16* Bb   = Bt + (long long)bz * sB;

  // B async-copy slots: 256 threads x 2 x 16B cover 64 rows x 128B
  const int crow = tid >> 2;                    // 0..63
  const int cseg = tid & 3;                     // 0..3 (16B each)
  const bf16* gsrcB = Bb + (size_t)(nt * 64 + crow) * K + cseg * 8;
  const unsigned ldsA = (unsigned)(size_t)(void*)Atile;
  const unsigned ldsB = (unsigned)(size_t)(void*)Btile + (unsigned)(crow * 128 + cseg * 16);

  // fragment base pointers into LDS (row stride = 64 elements)
  const bf16* Ald  = Atile + (size_t)(wave * 16 + lidx) * 64;
  const bf16* Bl0  = Btile + (size_t)( 0 + lidx) * 64;
  const bf16* Bl1  = Btile + (size_t)(16 + lidx) * 64;
  const bf16* Bl2  = Btile + (size_t)(32 + lidx) * 64;
  const bf16* Bl3  = Btile + (size_t)(48 + lidx) * 64;

  v8f acc0 = {}, acc1 = {}, acc2 = {}, acc3 = {};
  for (int k0 = 0; k0 < K; k0 += 64) {
    __syncthreads();                             // previous slab fully consumed
    if (wave == 0)
      tdm_load_2d(ldsA, Ablk + k0, /*tile_k=*/64, /*tile_rows=*/128,
                  (unsigned long long)K);        // TENSORcnt++
    async_copy_b128(ldsB,      gsrcB + k0);      // ASYNCcnt++
    async_copy_b128(ldsB + 64, gsrcB + k0 + 32); // ASYNCcnt++
    __builtin_prefetch(gsrcB + k0 + 128, 0, 0);  // global_prefetch_b8
    wait_async0();                               // s_wait_asynccnt 0
    if (wave == 0)
      __builtin_amdgcn_s_wait_tensorcnt(0);      // s_wait_tensorcnt 0
    __syncthreads();                             // slab visible to all waves

#pragma unroll
    for (int s = 0; s < 2; ++s) {                // two 16x16x32 k-substeps per slab
      bf16x16 a  = load_frag_a(Ald + s * 32, lane);   // ds_load_b128 x2
      bf16x16 b0 = load_frag_b(Bl0 + s * 32, lane);
      bf16x16 b1 = load_frag_b(Bl1 + s * 32, lane);
      bf16x16 b2 = load_frag_b(Bl2 + s * 32, lane);
      bf16x16 b3 = load_frag_b(Bl3 + s * 32, lane);
      acc0 = wmma_bf16(a, b0, acc0);
      acc1 = wmma_bf16(a, b1, acc1);
      acc2 = wmma_bf16(a, b2, acc2);
      acc3 = wmma_bf16(a, b3, acc3);
    }
  }

  OUT_T* Cb = C + (long long)bz * sC;
  const int mrow = mt * 16 + hlf * 8;    // C layout: VGPR j -> row j (+8 for upper half)
#pragma unroll
  for (int j = 0; j < 8; ++j) {
    OUT_T* p = Cb + (size_t)(mrow + j) * N + nt * 64 + lidx;
    store_val(p +  0, acc0[j]);
    store_val(p + 16, acc1[j]);
    store_val(p + 32, acc2[j]);
    store_val(p + 48, acc3[j]);
  }
}

// ---------------- softmax pass 1: per (b,h,q) row max & sum-exp (online) ----------------
// grid: (N_HEADS, SEQ/128, BATCH), block 256; wave -> 16 q-rows

__global__ __launch_bounds__(256) void k_attn_stats(
    const bf16* __restrict__ Q, const bf16* __restrict__ Kmat,
    float* __restrict__ m_out, float* __restrict__ s_out) {
  const int wave = threadIdx.x >> 5;
  const int lane = threadIdx.x & 31;
  const int hlf  = lane >> 4;
  const int lidx = lane & 15;
  const int bz = blockIdx.z;
  const int h  = blockIdx.x;
  const int qt = blockIdx.y * 8 + wave;     // 0..127

  const bf16* qrow = Q + ((size_t)(bz * SEQ + qt * 16 + lidx)) * D_MODEL + h * D_HEAD;

  float m_run[8], s_run[8];
#pragma unroll
  for (int j = 0; j < 8; ++j) { m_run[j] = -3.0e38f; s_run[j] = 0.f; }

  for (int kt = 0; kt <= qt; ++kt) {        // fully-masked tiles contribute exp()==0
    const bf16* krow = Kmat + ((size_t)(bz * SEQ + kt * 16 + lidx)) * D_MODEL + h * D_HEAD;
    v8f acc = {};
#pragma unroll
    for (int kk = 0; kk < 4; ++kk)          // d_head = 128 = 4 * 32
      acc = wmma_bf16(load_frag_a(qrow + kk * 32, lane),
                      load_frag_b(krow + kk * 32, lane), acc);

    const int colk = kt * 16 + lidx;
#pragma unroll
    for (int j = 0; j < 8; ++j) {
      const int rowq = qt * 16 + hlf * 8 + j;
      float sc = acc[j];
      if (colk > rowq) sc += -1.0e9f;       // quirk: mask BEFORE scale
      sc *= SCALE_F;
      float rmax  = hred_max(sc);
      float m_new = fmaxf(m_run[j], rmax);
      float e     = __expf(sc - m_new);
      float rsum  = hred_sum(e);
      s_run[j] = s_run[j] * __expf(m_run[j] - m_new) + rsum;
      m_run[j] = m_new;
    }
  }

  if (lidx == 0) {
#pragma unroll
    for (int j = 0; j < 8; ++j) {
      size_t idx = ((size_t)(bz * N_HEADS + h) * SEQ) + qt * 16 + hlf * 8 + j;
      m_out[idx] = m_run[j];
      s_out[idx] = s_run[j];
    }
  }
}

// ---------------- softmax pass 2: P_sum[b,q,k] = sum_h softmax(scores)[b,h,q,k] ----------------
// grid: (SEQ/16/8, SEQ/16, BATCH), block 256; wave -> one 16x16 (q,k) tile

__global__ __launch_bounds__(256) void k_psum(
    const bf16* __restrict__ Q, const bf16* __restrict__ Kmat,
    const float* __restrict__ m_in, const float* __restrict__ s_in,
    bf16* __restrict__ P) {
  const int wave = threadIdx.x >> 5;
  const int lane = threadIdx.x & 31;
  const int hlf  = lane >> 4;
  const int lidx = lane & 15;
  const int bz = blockIdx.z;
  const int qt = blockIdx.y;
  const int kt = blockIdx.x * 8 + wave;

  bf16* Pb = P + (size_t)bz * SEQ * SEQ;
  const int colk = kt * 16 + lidx;

  if (kt > qt) {                             // fully masked: probs == 0 (must write; ws is poisoned)
#pragma unroll
    for (int j = 0; j < 8; ++j) {
      const int rowq = qt * 16 + hlf * 8 + j;
      Pb[(size_t)rowq * SEQ + colk] = f2bf(0.f);
    }
    return;
  }

  float psum[8];
#pragma unroll
  for (int j = 0; j < 8; ++j) psum[j] = 0.f;

  for (int h = 0; h < N_HEADS; ++h) {
    const bf16* qrow = Q    + ((size_t)(bz * SEQ + qt * 16 + lidx)) * D_MODEL + h * D_HEAD;
    const bf16* krow = Kmat + ((size_t)(bz * SEQ + kt * 16 + lidx)) * D_MODEL + h * D_HEAD;
    v8f acc = {};
#pragma unroll
    for (int kk = 0; kk < 4; ++kk)
      acc = wmma_bf16(load_frag_a(qrow + kk * 32, lane),
                      load_frag_b(krow + kk * 32, lane), acc);

    const float* mrow = m_in + (size_t)(bz * N_HEADS + h) * SEQ + qt * 16 + hlf * 8;
    const float* srow = s_in + (size_t)(bz * N_HEADS + h) * SEQ + qt * 16 + hlf * 8;
#pragma unroll
    for (int j = 0; j < 8; ++j) {
      const int rowq = qt * 16 + hlf * 8 + j;
      float sc = acc[j];
      if (colk > rowq) sc += -1.0e9f;
      sc *= SCALE_F;
      psum[j] += __expf(sc - mrow[j]) / srow[j];
    }
  }

#pragma unroll
  for (int j = 0; j < 8; ++j) {
    const int rowq = qt * 16 + hlf * 8 + j;
    Pb[(size_t)rowq * SEQ + colk] = f2bf(psum[j]);
  }
}

// ---------------- launcher ----------------

extern "C" void kernel_launch(void* const* d_in, const int* in_sizes, int n_in,
                              void* d_out, int out_size, void* d_ws, size_t ws_size,
                              hipStream_t stream) {
  (void)in_sizes; (void)n_in; (void)out_size; (void)ws_size;
  const float* x  = (const float*)d_in[0];
  const float* Wq = (const float*)d_in[1];
  const float* Wk = (const float*)d_in[2];
  const float* Wv = (const float*)d_in[3];
  const float* Wo = (const float*)d_in[4];
  float* out = (float*)d_out;

  char* w = (char*)d_ws;
  auto alloc = [&](size_t bytes) {
    char* p = w;
    w += (bytes + 255) & ~(size_t)255;
    return p;
  };
  const size_t XE = (size_t)TOKENS * D_MODEL;    // 8,388,608 elems (also B*S*S)
  const size_t WE = (size_t)D_MODEL * D_MODEL;   // 4,194,304 elems

  bf16* x16  = (bf16*)alloc(XE * 2);
  bf16* Wtq  = (bf16*)alloc(WE * 2);
  bf16* Wtk  = (bf16*)alloc(WE * 2);
  bf16* Wtv  = (bf16*)alloc(WE * 2);
  bf16* Wto  = (bf16*)alloc(WE * 2);
  bf16* Q16  = (bf16*)alloc(XE * 2);
  bf16* K16  = (bf16*)alloc(XE * 2);
  bf16* V16  = (bf16*)alloc(XE * 2);
  bf16* Vt16 = (bf16*)alloc(XE * 2);
  bf16* Ps   = (bf16*)alloc(XE * 2);             // P_sum: [B][S][S]
  bf16* At   = (bf16*)alloc(XE * 2);             // attn:  [B][S][D_MODEL]
  float* m_a = (float*)alloc((size_t)BATCH * N_HEADS * SEQ * 4);
  float* s_a = (float*)alloc((size_t)BATCH * N_HEADS * SEQ * 4);

  dim3 blk(256);

  // 1) x -> bf16
  k_f32_to_bf16<<<dim3(2048), blk, 0, stream>>>(x, x16, (long long)XE);

  // 2) W -> W^T bf16
  dim3 tg(D_MODEL / 32, D_MODEL / 32, 1);
  k_transpose_f32_bf16<<<tg, blk, 0, stream>>>(Wq, Wtq, D_MODEL, D_MODEL);
  k_transpose_f32_bf16<<<tg, blk, 0, stream>>>(Wk, Wtk, D_MODEL, D_MODEL);
  k_transpose_f32_bf16<<<tg, blk, 0, stream>>>(Wv, Wtv, D_MODEL, D_MODEL);
  k_transpose_f32_bf16<<<tg, blk, 0, stream>>>(Wo, Wto, D_MODEL, D_MODEL);

  // 3) Q,K,V projections (M = 4096 tokens)
  dim3 g1(D_MODEL / 64, TOKENS / 128, 1);
  k_gemm_bf16<bf16><<<g1, blk, 0, stream>>>(x16, Wtq, Q16, D_MODEL, D_MODEL, 0, 0, 0);
  k_gemm_bf16<bf16><<<g1, blk, 0, stream>>>(x16, Wtk, K16, D_MODEL, D_MODEL, 0, 0, 0);
  k_gemm_bf16<bf16><<<g1, blk, 0, stream>>>(x16, Wtv, V16, D_MODEL, D_MODEL, 0, 0, 0);

  // 4) V -> V^T per batch (so PV GEMM reads contiguous K)
  dim3 tg2(D_MODEL / 32, SEQ / 32, BATCH);
  k_transpose_bf16<<<tg2, blk, 0, stream>>>(V16, Vt16, SEQ, D_MODEL);

  // 5) softmax stats (online max/sumexp per (b,h,q))
  k_attn_stats<<<dim3(N_HEADS, SEQ / 128, BATCH), blk, 0, stream>>>(Q16, K16, m_a, s_a);

  // 6) head-summed probabilities P_sum
  k_psum<<<dim3(SEQ / 16 / 8, SEQ / 16, BATCH), blk, 0, stream>>>(Q16, K16, m_a, s_a, Ps);

  // 7) attn = P_sum @ V   (batched, K = SEQ)
  dim3 g2(D_MODEL / 64, SEQ / 128, BATCH);
  k_gemm_bf16<bf16><<<g2, blk, 0, stream>>>(
      Ps, Vt16, At, SEQ, D_MODEL,
      (long long)SEQ * SEQ, (long long)SEQ * D_MODEL, (long long)SEQ * D_MODEL);

  // 8) out = attn @ W_o  (f32 output)
  k_gemm_bf16<float><<<g1, blk, 0, stream>>>(At, Wto, out, D_MODEL, D_MODEL, 0, 0, 0);
}
